// GINE_3831110828525
// MI455X (gfx1250) — matrix operand order repaired and checked by reading, not compiled, not run on hardware
//
#include <hip/hip_runtime.h>
#include <hip/hip_bf16.h>

// MI455X / gfx1250: wave32, WMMA 16x16x32 bf16 (f32 accumulate).
typedef __attribute__((ext_vector_type(16))) __bf16 v16bf;
typedef __attribute__((ext_vector_type(8)))  float  v8f;
typedef __attribute__((ext_vector_type(4)))  float  v4f;   // native vector (NT-load capable)

#define N_NODES 50000
#define N_EDGES 800000
#define HID     128
#define H2      256
#define OUTC    64
#define NLAYERS 3

// ---------------------------------------------------------------------------
// Weight pre-swizzle: row-major fp32 [K x NCOL] -> bf16 WMMA B-fragments.
// B-frag layout used here (contiguous k, same order as A): lane holds column n = lane&15,
// element j holds k = ks*32 + j + 16*(lane>>4). One 32-byte load per frag.
// ---------------------------------------------------------------------------
__global__ void prep_wfrag(const float* __restrict__ W, __bf16* __restrict__ out,
                           int K, int NCOL) {
  const int tid = blockIdx.x * blockDim.x + threadIdx.x;
  const int KS = K >> 5;
  const int total = (NCOL >> 4) * KS * 32;
  if (tid >= total) return;
  const int lane = tid & 31;
  const int fi = tid >> 5;
  const int ks = fi % KS;
  const int nt = fi / KS;
  const int h = lane >> 4, ln = lane & 15;
  __bf16* o = out + (size_t)tid * 16;
#pragma unroll
  for (int j = 0; j < 16; ++j) {
    const int k = ks * 32 + j + 16 * h;
    o[j] = (__bf16)W[(size_t)k * NCOL + nt * 16 + ln];
  }
}

// ---------------------------------------------------------------------------
// Fused bf16-WMMA GEMM. Each wave owns a 16x64 output tile (4 ntiles,
// 32 acc VGPRs). A-frag layout (16-bit A 16x32): lane row m = lane&15,
// k = j + 16*(lane>>4), built on the fly from fp32 with a
// mode-specific prologue. KSV is a compile-time K/32 so the K loop fully
// unrolls and B-fragment loads get hoisted ahead of the WMMA chain.
// ---------------------------------------------------------------------------
enum { M_PLAIN = 0, M_COMBINE = 1, M_BNRELU = 2 };

template<int MODE>
__device__ __forceinline__ float fetchA(const float* __restrict__ A0,
                                        const float* __restrict__ A1,
                                        const float* __restrict__ ss,
                                        size_t ro, int k, int K, float alpha) {
  if constexpr (MODE == M_PLAIN) {
    return A0[ro + k];
  } else if constexpr (MODE == M_COMBINE) {
    return alpha * A0[ro + k] + A1[ro + k];          // (1+eps)*x + agg
  } else {
    const float v = A0[ro + k];
    return fmaxf(0.0f, v * ss[k] + ss[K + k]);       // relu(BN(h1))
  }
}

template<int MODE, bool STATS, int KSV, bool NTSTORE>
__global__ void gine_gemm(const float* __restrict__ A0, const float* __restrict__ A1,
                          const float* __restrict__ ss, const __bf16* __restrict__ Bfrag,
                          const float* __restrict__ bias, const float* __restrict__ epsArr,
                          int layer, float* __restrict__ C, float* __restrict__ stats,
                          int M, int NCOL) {
  const int K = KSV * 32;
  const int mtiles = M >> 4;
  const int ngroups = NCOL >> 6;     // 64 output cols per wave
  const int wave = blockIdx.x * (blockDim.x >> 5) + (threadIdx.x >> 5);
  if (wave >= mtiles * ngroups) return;   // wave-uniform: EXEC stays all-ones
  const int mtile = wave % mtiles;
  const int ng = wave / mtiles;
  const int lane = threadIdx.x & 31;
  const int h = lane >> 4, ln = lane & 15;
  const size_t ro = (size_t)(mtile * 16 + ln) * K;

  float alpha = 1.0f;
  if constexpr (MODE == M_COMBINE) alpha = 1.0f + epsArr[layer];

  // B fragments for this wave start at ntile group ng*4.
  const __bf16* bbase = Bfrag + (size_t)(ng * 4) * KSV * 32 * 16;

  v8f acc[4];
#pragma unroll
  for (int nt = 0; nt < 4; ++nt)
#pragma unroll
    for (int v = 0; v < 8; ++v) acc[nt][v] = 0.0f;

#pragma unroll
  for (int ks = 0; ks < KSV; ++ks) {
    // B fragments first: with the loop fully unrolled the scheduler can issue
    // these well ahead of the consuming WMMAs.
    v16bf b[4];
#pragma unroll
    for (int nt = 0; nt < 4; ++nt)
      b[nt] = *(const v16bf*)(bbase + ((size_t)(nt * KSV + ks) * 32 + lane) * 16);

    // A fragment (fp32 -> bf16 with fused prologue).
    const int kb = ks * 32 + 16 * h;
    float f[16];
#pragma unroll
    for (int j = 0; j < 8; ++j) f[j]     = fetchA<MODE>(A0, A1, ss, ro, kb + j,      K, alpha);
#pragma unroll
    for (int j = 0; j < 8; ++j) f[8 + j] = fetchA<MODE>(A0, A1, ss, ro, kb + 8 + j, K, alpha);
    v16bf a;
#pragma unroll
    for (int j = 0; j < 16; ++j) a[j] = (__bf16)f[j];

#pragma unroll
    for (int nt = 0; nt < 4; ++nt)
      acc[nt] = __builtin_amdgcn_wmma_f32_16x16x32_bf16(false, a, false, b[nt],
                                                        (short)0, acc[nt], false, false);
  }

  // Epilogue: bias add, store, optional BN-stat accumulation.
#pragma unroll
  for (int nt = 0; nt < 4; ++nt) {
    const int col = (ng * 4 + nt) * 16 + ln;
    const float bi = bias[col];
    float s = 0.0f, q = 0.0f;
#pragma unroll
    for (int v = 0; v < 8; ++v) {
      const float val = acc[nt][v] + bi;             // C/D layout: m = v + 8*h
      float* cp = &C[(size_t)(mtile * 16 + v + 8 * h) * NCOL + col];
      if constexpr (NTSTORE) __builtin_nontemporal_store(val, cp);
      else                   *cp = val;
      if constexpr (STATS) { s += val; q += val * val; }
    }
    if constexpr (STATS) {
      s += __shfl_xor(s, 16, 32);                    // combine the two half-waves
      q += __shfl_xor(q, 16, 32);
      if (lane < 16) {
        atomicAdd(&stats[col], s);
        atomicAdd(&stats[NCOL + col], q);
      }
    }
  }
}

// ---------------------------------------------------------------------------
// Edge phase: msg = relu(x[src] + e), atomic scatter-add into agg[dst].
// One thread per (edge, 4-float chunk). e (409.6 MB) is a pure stream ->
// non-temporal loads so it does not evict the L2-resident x/agg (25.6 MB).
// ---------------------------------------------------------------------------
__global__ void edge_message(const float* __restrict__ x, const float* __restrict__ e,
                             const int* __restrict__ ei, float* __restrict__ agg,
                             int E) {
  const long long t = (long long)blockIdx.x * blockDim.x + threadIdx.x;
  if (t >= (long long)E * 32) return;
  const int edge = (int)(t >> 5);
  const int c = (int)(t & 31) * 4;
  const int src = ei[edge];
  const int dst = ei[E + edge];
  const v4f ev = __builtin_nontemporal_load((const v4f*)(e + (size_t)edge * HID + c));
  const v4f xv = *(const v4f*)(x + (size_t)src * HID + c);
  float* ag = agg + (size_t)dst * HID + c;
  atomicAdd(ag + 0, fmaxf(0.0f, ev.x + xv.x));
  atomicAdd(ag + 1, fmaxf(0.0f, ev.y + xv.y));
  atomicAdd(ag + 2, fmaxf(0.0f, ev.z + xv.z));
  atomicAdd(ag + 3, fmaxf(0.0f, ev.w + xv.w));
}

// Turn (sum, sumsq) into per-channel (scale, shift): y = h*scale + shift.
__global__ void bn_finalize(const float* __restrict__ stats, const float* __restrict__ g,
                            const float* __restrict__ b, float* __restrict__ ss,
                            int NCOL, float invN) {
  const int c = blockIdx.x * blockDim.x + threadIdx.x;
  if (c >= NCOL) return;
  const float mu = stats[c] * invN;
  const float var = stats[NCOL + c] * invN - mu * mu;
  const float sc = g[c] * rsqrtf(var + 1e-5f);
  ss[c] = sc;
  ss[NCOL + c] = b[c] - mu * sc;
}

// x = relu(BN(h2))
__global__ void x_update(const float* __restrict__ h2, const float* __restrict__ ss,
                         float* __restrict__ x, int N) {
  const long long i = (long long)blockIdx.x * blockDim.x + threadIdx.x;
  if (i >= (long long)N * HID) return;
  const int c = (int)(i & (HID - 1));
  x[i] = fmaxf(0.0f, h2[i] * ss[c] + ss[HID + c]);
}

// ---------------------------------------------------------------------------
extern "C" void kernel_launch(void* const* d_in, const int* in_sizes, int n_in,
                              void* d_out, int out_size, void* d_ws, size_t ws_size,
                              hipStream_t stream) {
  (void)in_sizes; (void)n_in; (void)out_size; (void)ws_size;
  const float* x_in    = (const float*)d_in[0];
  const float* eattr   = (const float*)d_in[1];
  const float* atom_w  = (const float*)d_in[2];
  const float* atom_b  = (const float*)d_in[3];
  const float* bond_w  = (const float*)d_in[4];
  const float* bond_b  = (const float*)d_in[5];
  const float* epsArr  = (const float*)d_in[6];
  const float* lin1_w  = (const float*)d_in[7];
  const float* lin1_b  = (const float*)d_in[8];
  const float* bn1_g   = (const float*)d_in[9];
  const float* bn1_b   = (const float*)d_in[10];
  const float* lin2_w  = (const float*)d_in[11];
  const float* lin2_b  = (const float*)d_in[12];
  const float* norm_g  = (const float*)d_in[13];
  const float* norm_b  = (const float*)d_in[14];
  const float* lin_w   = (const float*)d_in[15];
  const float* lin_b   = (const float*)d_in[16];
  const int*   eidx    = (const int*)d_in[17];   // JAX default: int32
  float* out = (float*)d_out;

  // ---- workspace carve-up (256-B aligned) ----
  char* ws = (char*)d_ws;
  size_t off = 0;
  auto carve = [&](size_t bytes) -> char* {
    char* p = ws + off;
    off += (bytes + 255) & ~(size_t)255;
    return p;
  };
  float*  e     = (float*)carve((size_t)N_EDGES * HID * 4);   // 409.6 MB
  float*  x_cur = (float*)carve((size_t)N_NODES * HID * 4);
  float*  agg   = (float*)carve((size_t)N_NODES * HID * 4);
  float*  h1    = (float*)carve((size_t)N_NODES * H2  * 4);
  float*  h2    = (float*)carve((size_t)N_NODES * HID * 4);
  float*  st1   = (float*)carve(2 * H2  * 4);
  float*  st2   = (float*)carve(2 * HID * 4);
  float*  ss1   = (float*)carve(2 * H2  * 4);
  float*  ss2   = (float*)carve(2 * HID * 4);
  __bf16* f_atom = (__bf16*)carve((size_t)32  * HID * 2);
  __bf16* f_bond = (__bf16*)carve((size_t)32  * HID * 2);
  __bf16* f_lin1 = (__bf16*)carve((size_t)NLAYERS * HID * H2  * 2);
  __bf16* f_lin2 = (__bf16*)carve((size_t)NLAYERS * H2  * HID * 2);
  __bf16* f_lin  = (__bf16*)carve((size_t)HID * OUTC * 2);

  // ---- weight pre-swizzle into WMMA B-fragment order ----
  prep_wfrag<<<1, 256, 0, stream>>>(atom_w, f_atom, 32, HID);
  prep_wfrag<<<1, 256, 0, stream>>>(bond_w, f_bond, 32, HID);
  for (int i = 0; i < NLAYERS; ++i) {
    prep_wfrag<<<8, 256, 0, stream>>>(lin1_w + (size_t)i * HID * H2,
                                      f_lin1 + (size_t)i * HID * H2, HID, H2);
    prep_wfrag<<<8, 256, 0, stream>>>(lin2_w + (size_t)i * H2 * HID,
                                      f_lin2 + (size_t)i * H2 * HID, H2, HID);
  }
  prep_wfrag<<<2, 256, 0, stream>>>(lin_w, f_lin, HID, OUTC);

  const int TPB = 256;  // 8 waves per block
  auto gblocks = [](int M, int NCOL) { return ((M >> 4) * (NCOL >> 6) + 7) / 8; };

  // ---- node / edge embeddings (K=32 -> KSV=1) ----
  gine_gemm<M_PLAIN, false, 1, false><<<gblocks(N_NODES, HID), TPB, 0, stream>>>(
      x_in, nullptr, nullptr, f_atom, atom_b, nullptr, 0,
      x_cur, nullptr, N_NODES, HID);
  gine_gemm<M_PLAIN, false, 1, true><<<gblocks(N_EDGES, HID), TPB, 0, stream>>>(
      eattr, nullptr, nullptr, f_bond, bond_b, nullptr, 0,
      e, nullptr, N_EDGES, HID);   // NT store: e streams past L2

  const float invN = 1.0f / (float)N_NODES;
  const long long msg_threads = (long long)N_EDGES * 32;
  const int msg_blocks = (int)((msg_threads + 255) / 256);
  const int upd_blocks = (int)(((long long)N_NODES * HID + 255) / 256);

  for (int i = 0; i < NLAYERS; ++i) {
    // message + scatter-sum
    (void)hipMemsetAsync(agg, 0, (size_t)N_NODES * HID * 4, stream);
    edge_message<<<msg_blocks, 256, 0, stream>>>(x_cur, e, eidx, agg, N_EDGES);

    // h1 = ((1+eps)x + agg) @ lin1_w + lin1_b   (+ BN1 stats), K=128 -> KSV=4
    (void)hipMemsetAsync(st1, 0, 2 * H2 * 4, stream);
    gine_gemm<M_COMBINE, true, 4, false><<<gblocks(N_NODES, H2), TPB, 0, stream>>>(
        x_cur, agg, nullptr, f_lin1 + (size_t)i * HID * H2, lin1_b + i * H2,
        epsArr, i, h1, st1, N_NODES, H2);
    bn_finalize<<<1, H2, 0, stream>>>(st1, bn1_g + i * H2, bn1_b + i * H2, ss1, H2, invN);

    // h2 = relu(BN1(h1)) @ lin2_w + lin2_b      (+ BN2 stats), K=256 -> KSV=8
    (void)hipMemsetAsync(st2, 0, 2 * HID * 4, stream);
    gine_gemm<M_BNRELU, true, 8, false><<<gblocks(N_NODES, HID), TPB, 0, stream>>>(
        h1, nullptr, ss1, f_lin2 + (size_t)i * H2 * HID, lin2_b + i * HID,
        nullptr, 0, h2, st2, N_NODES, HID);
    bn_finalize<<<1, HID, 0, stream>>>(st2, norm_g + i * HID, norm_b + i * HID, ss2, HID, invN);

    // x = relu(BN2(h2))
    x_update<<<upd_blocks, 256, 0, stream>>>(h2, ss2, x_cur, N_NODES);
  }

  // out = x @ lin_w + lin_b  (K=128 -> KSV=4)
  gine_gemm<M_PLAIN, false, 4, false><<<gblocks(N_NODES, OUTC), TPB, 0, stream>>>(
      x_cur, nullptr, nullptr, f_lin, lin_b, nullptr, 0,
      out, nullptr, N_NODES, OUTC);
}
